// GPT_70841190580594
// MI455X (gfx1250) — matrix operand order repaired and checked by reading, not compiled, not run on hardware
//
#include <hip/hip_runtime.h>
#include <math.h>

// ---------------- GPT-2 small constants ----------------
#define LYR 12
#define HDS 12
#define CH  768
#define VOC 50257
#define SEQ 1024
#define BAT 4
#define DH  64

typedef __attribute__((ext_vector_type(16))) __bf16 v16bf;
typedef __attribute__((ext_vector_type(2)))  __bf16 v2bf;
typedef __attribute__((ext_vector_type(8)))  float  v8f;
typedef __attribute__((ext_vector_type(4)))  unsigned int u32x4;

// ---------------- helpers ----------------
__device__ __forceinline__ unsigned int pk2bf(float a, float b) {
    union { v2bf h; unsigned int u; } x;
    v2bf h; h[0] = (__bf16)a; h[1] = (__bf16)b;
    x.h = h;
    return x.u;
}
__device__ __forceinline__ unsigned short bf1(float a) {
    union { __bf16 h; unsigned short u; } x;
    x.h = (__bf16)a;
    return x.u;
}
__device__ __forceinline__ v16bf ld_frag(const unsigned short* p0, const unsigned short* p1) {
    union { u32x4 q[2]; v16bf v; } u;
    u.q[0] = *(const u32x4*)p0;
    u.q[1] = *(const u32x4*)p1;
    return u.v;
}
__device__ __forceinline__ float gelu_tanh(float v) {
    const float c = 0.7978845608028654f;
    return 0.5f * v * (1.0f + tanhf(c * (v + 0.044715f * v * v * v)));
}

// ---------------- WMMA GEMM: C[M,N] = A[M,K] @ B + bias (+gelu) (+resid) -----
// BT=false: B is [K,N] row-major (weights).  BT=true: B is [N,K] row-major (lm_head).
// Block tile 128x128, K-step 32, 256 threads = 8 wave32 waves.
// Each wave: 32(M) x 64(N) = 2x4 tiles of v_wmma_f32_16x16x32_bf16.
// Double-buffered LDS; next tile's loads issued before this tile's WMMAs;
// global_prefetch two tiles ahead.
#define TM 128
#define TN 128
#define TKS 32
#define PIT 40   // LDS row pitch in bf16 elements (multiple of 8 -> 16B-aligned chunks)

template<bool BT, bool GELU>
__global__ __launch_bounds__(256) void gemm_wmma_bf16(
    const float* __restrict__ A, const float* __restrict__ Bm,
    const float* __restrict__ bias, const float* __restrict__ resid,
    float* __restrict__ Cout, int M, int N, int K)
{
    __shared__ unsigned short lA[2][TM * PIT];  // [m][k]
    __shared__ unsigned short lB[2][TN * PIT];  // [n][k] (B transposed into LDS)

    const int tid  = threadIdx.x;
    const int lane = tid & 31;
    const int wv   = tid >> 5;
    const int wm   = wv & 3;
    const int wn   = wv >> 2;
    const int bM   = blockIdx.y * TM;
    const int bN   = blockIdx.x * TN;
    const int m16  = lane & 15;
    const int half = lane >> 4;

    const int am0 = tid >> 3;
    const int akq = (tid & 7) << 2;
    const int bkk0 = tid >> 5;
    const int bnq  = (tid & 31) << 2;

    v8f acc[2][4];
#pragma unroll
    for (int i = 0; i < 2; ++i)
#pragma unroll
        for (int j = 0; j < 4; ++j) acc[i][j] = (v8f){0.f,0.f,0.f,0.f,0.f,0.f,0.f,0.f};

    float4 aReg[4], bReg[4];

    auto load_regs = [&](int k0) {
#pragma unroll
        for (int i = 0; i < 4; ++i)
            aReg[i] = *(const float4*)&A[(size_t)(bM + am0 + 32 * i) * K + k0 + akq];
        if constexpr (BT) {
#pragma unroll
            for (int i = 0; i < 4; ++i) {
                int n = am0 + 32 * i;
                bReg[i] = make_float4(0.f, 0.f, 0.f, 0.f);
                if (bN + n < N)
                    bReg[i] = *(const float4*)&Bm[(size_t)(bN + n) * K + k0 + akq];
            }
        } else {
#pragma unroll
            for (int i = 0; i < 4; ++i)
                bReg[i] = *(const float4*)&Bm[(size_t)(k0 + bkk0 + 8 * i) * N + bN + bnq];
        }
    };

    auto prefetch_tile = [&](int k0) {
#pragma unroll
        for (int i = 0; i < 4; ++i)
            __builtin_prefetch(&A[(size_t)(bM + am0 + 32 * i) * K + k0 + akq], 0, 1);
        if constexpr (BT) {
#pragma unroll
            for (int i = 0; i < 4; ++i)
                if (bN + am0 + 32 * i < N)
                    __builtin_prefetch(&Bm[(size_t)(bN + am0 + 32 * i) * K + k0 + akq], 0, 1);
        } else {
#pragma unroll
            for (int i = 0; i < 4; ++i)
                __builtin_prefetch(&Bm[(size_t)(k0 + bkk0 + 8 * i) * N + bN + bnq], 0, 1);
        }
    };

    auto cvt_store = [&](int buf) {
#pragma unroll
        for (int i = 0; i < 4; ++i) {
            uint2 p; p.x = pk2bf(aReg[i].x, aReg[i].y); p.y = pk2bf(aReg[i].z, aReg[i].w);
            *(uint2*)&lA[buf][(am0 + 32 * i) * PIT + akq] = p;
        }
        if constexpr (BT) {
#pragma unroll
            for (int i = 0; i < 4; ++i) {
                uint2 p; p.x = pk2bf(bReg[i].x, bReg[i].y); p.y = pk2bf(bReg[i].z, bReg[i].w);
                *(uint2*)&lB[buf][(am0 + 32 * i) * PIT + akq] = p;
            }
        } else {
#pragma unroll
            for (int i = 0; i < 4; ++i) {
                int kk = bkk0 + 8 * i;
                lB[buf][(bnq + 0) * PIT + kk] = bf1(bReg[i].x);
                lB[buf][(bnq + 1) * PIT + kk] = bf1(bReg[i].y);
                lB[buf][(bnq + 2) * PIT + kk] = bf1(bReg[i].z);
                lB[buf][(bnq + 3) * PIT + kk] = bf1(bReg[i].w);
            }
        }
    };

    load_regs(0);
    cvt_store(0);
    __syncthreads();

    const int nIter = K / TKS;
    for (int it = 0; it < nIter; ++it) {
        const int cur = it & 1;
        const bool more = (it + 1 < nIter);
        if (more) load_regs((it + 1) * TKS);
        if (it + 2 < nIter) prefetch_tile((it + 2) * TKS);

        v16bf af[2], bfr[4];
#pragma unroll
        for (int i = 0; i < 2; ++i) {
            int base = (wm * 32 + i * 16 + m16) * PIT + 8 * half;
            af[i] = ld_frag(&lA[cur][base], &lA[cur][base + 16]);
        }
#pragma unroll
        for (int j = 0; j < 4; ++j) {
            int base = (wn * 64 + j * 16 + m16) * PIT + 8 * half;
            bfr[j] = ld_frag(&lB[cur][base], &lB[cur][base + 16]);
        }
#pragma unroll
        for (int i = 0; i < 2; ++i)
#pragma unroll
            for (int j = 0; j < 4; ++j)
                acc[i][j] = __builtin_amdgcn_wmma_f32_16x16x32_bf16(
                    false, af[i], false, bfr[j], (short)0, acc[i][j], false, false);

        if (more) cvt_store(1 - cur);
        __syncthreads();
    }

#pragma unroll
    for (int i = 0; i < 2; ++i) {
        int row0 = bM + wm * 32 + i * 16 + half * 8;
#pragma unroll
        for (int j = 0; j < 4; ++j) {
            int col = bN + wn * 64 + j * 16 + m16;
            if (col < N) {
                float bb = bias ? bias[col] : 0.f;
#pragma unroll
                for (int r = 0; r < 8; ++r) {
                    float v = acc[i][j][r] + bb;
                    if (GELU) v = gelu_tanh(v);
                    size_t off = (size_t)(row0 + r) * N + col;
                    if (resid) v += resid[off];
                    Cout[off] = v;
                }
            }
        }
    }
}

// ---------------- embedding ----------------
__global__ __launch_bounds__(256) void embed_k(const int* __restrict__ idx,
                                               const float* __restrict__ wte,
                                               const float* __restrict__ wpe,
                                               float* __restrict__ x)
{
    int row = blockIdx.x;
    int t   = row & (SEQ - 1);
    int tok = idx[row];
    const float* we = wte + (size_t)tok * CH;
    const float* pe = wpe + (size_t)t * CH;
    float* xr = x + (size_t)row * CH;
    for (int c = threadIdx.x; c < CH; c += 256)
        xr[c] = we[c] + pe[c];
}

// ---------------- layernorm (block per row) ----------------
__global__ __launch_bounds__(256) void layernorm_k(const float* __restrict__ x,
                                                   const float* __restrict__ w,
                                                   const float* __restrict__ b,
                                                   float* __restrict__ out)
{
    __shared__ float s1[256], s2[256];
    int row = blockIdx.x;
    int tid = threadIdx.x;
    const float* xr = x + (size_t)row * CH;
    float* orow = out + (size_t)row * CH;

    float a = 0.f, a2 = 0.f;
    for (int c = tid; c < CH; c += 256) { float v = xr[c]; a += v; a2 += v * v; }
    s1[tid] = a; s2[tid] = a2;
    __syncthreads();
    for (int o = 128; o > 0; o >>= 1) {
        if (tid < o) { s1[tid] += s1[tid + o]; s2[tid] += s2[tid + o]; }
        __syncthreads();
    }
    float mean = s1[0] * (1.0f / CH);
    float var  = s2[0] * (1.0f / CH) - mean * mean;
    float rstd = rsqrtf(var + 1e-5f);
    for (int c = tid; c < CH; c += 256)
        orow[c] = (xr[c] - mean) * rstd * w[c] + b[c];
}

// ---------------- flash attention on WMMA ----------------
// Block = (32-query tile, b*H+h). 128 threads = 4 waves.
// Wave w processes key tiles kt = w, w+4, ... with private online-softmax state;
// partials merged at the end (flash combine).
#define QT 32
#define KTL 32
#define QP 72    // q LDS pitch   [m][d]
#define KP 72    // k LDS pitch   [j][d]
#define VP 40    // v LDS pitch   [d][j]
#define PP 40    // p LDS pitch   [m][k]

__global__ __launch_bounds__(128) void attn_wmma_k(const float* __restrict__ qkv,
                                                   float* __restrict__ y)
{
    const int qt = blockIdx.x;            // query tile 0..SEQ/32-1
    const int bh = blockIdx.y;            // b*HDS + h
    const int hh = bh % HDS;
    const int b  = bh / HDS;

    const int tid  = threadIdx.x;
    const int lane = tid & 31;
    const int w    = tid >> 5;
    const int m16  = lane & 15;
    const int half = lane >> 4;

    __shared__ unsigned short lq[QT * QP];
    __shared__ unsigned short lk[4][KTL * KP];
    __shared__ unsigned short lv[4][DH * VP];
    __shared__ unsigned short lp[4][QT * PP];
    __shared__ float sY[4][QT][DH];
    __shared__ float sM[4][QT];
    __shared__ float sS[4][QT];
    __shared__ float sFac[4][QT];
    __shared__ float sDinv[QT];

    const float scale = 0.125f;           // 1/sqrt(64)
    const int rs = 3 * CH;

    // ---- stage q tile (32x64) as scaled bf16 ----
    {
        int c = tid;
#pragma unroll
        for (int i = 0; i < 4; ++i, c += 128) {
            int m  = c >> 4;              // 16 float4 per row
            int dq = (c & 15) << 2;
            const float4 f = *(const float4*)&qkv[(size_t)(b * SEQ + qt * QT + m) * rs + hh * DH + dq];
            uint2 p; p.x = pk2bf(f.x * scale, f.y * scale); p.y = pk2bf(f.z * scale, f.w * scale);
            *(uint2*)&lq[m * QP + dq] = p;
        }
    }
    __syncthreads();

    // q A-fragments (once): [ti: M tile][ks: K 32-chunk]
    v16bf qf[2][2];
#pragma unroll
    for (int ti = 0; ti < 2; ++ti)
#pragma unroll
        for (int ks = 0; ks < 2; ++ks) {
            int base = (ti * 16 + m16) * QP + ks * 32 + 8 * half;
            qf[ti][ks] = ld_frag(&lq[base], &lq[base + 16]);
        }

    v8f yac[2][4];
#pragma unroll
    for (int i = 0; i < 2; ++i)
#pragma unroll
        for (int j = 0; j < 4; ++j) yac[i][j] = (v8f){0.f,0.f,0.f,0.f,0.f,0.f,0.f,0.f};
    float mrun[2][8], srun[2][8];
#pragma unroll
    for (int ti = 0; ti < 2; ++ti)
#pragma unroll
        for (int r = 0; r < 8; ++r) { mrun[ti][r] = -1e30f; srun[ti][r] = 0.f; }

    const int nkt = qt + 1;
    for (int kt = w; kt < nkt; kt += 4) {
        // ---- stage K tile [j][d] and V tile [d][j] (bf16, wave-private) ----
        {
            const float* kr = &qkv[(size_t)(b * SEQ + kt * KTL + lane) * rs + CH + hh * DH];
#pragma unroll
            for (int dq = 0; dq < DH; dq += 4) {
                float4 f = *(const float4*)&kr[dq];
                uint2 p; p.x = pk2bf(f.x, f.y); p.y = pk2bf(f.z, f.w);
                *(uint2*)&lk[w][lane * KP + dq] = p;
            }
            const float* vr = &qkv[(size_t)(b * SEQ + kt * KTL + lane) * rs + 2 * CH + hh * DH];
#pragma unroll
            for (int d = 0; d < DH; ++d)
                lv[w][d * VP + lane] = bf1(vr[d]);
        }
        // same-wave LDS ops are in-order: no barrier needed (wave-private buffers)

        // ---- S = Q @ K^T : 2x2 tiles, K=64 ----
        v8f s[2][2];
#pragma unroll
        for (int i = 0; i < 2; ++i)
#pragma unroll
            for (int j = 0; j < 2; ++j) s[i][j] = (v8f){0.f,0.f,0.f,0.f,0.f,0.f,0.f,0.f};
#pragma unroll
        for (int ks = 0; ks < 2; ++ks) {
            v16bf kf[2];
#pragma unroll
            for (int tj = 0; tj < 2; ++tj) {
                int base = (tj * 16 + m16) * KP + ks * 32 + 8 * half;
                kf[tj] = ld_frag(&lk[w][base], &lk[w][base + 16]);
            }
#pragma unroll
            for (int ti = 0; ti < 2; ++ti)
#pragma unroll
                for (int tj = 0; tj < 2; ++tj)
                    s[ti][tj] = __builtin_amdgcn_wmma_f32_16x16x32_bf16(
                        false, qf[ti][ks], false, kf[tj], (short)0, s[ti][tj], false, false);
        }

        // ---- causal mask on diagonal tile ----
        if (kt == qt) {
#pragma unroll
            for (int ti = 0; ti < 2; ++ti)
#pragma unroll
                for (int tj = 0; tj < 2; ++tj)
#pragma unroll
                    for (int r = 0; r < 8; ++r) {
                        int mloc = ti * 16 + half * 8 + r;
                        int nloc = tj * 16 + m16;
                        if (nloc > mloc) s[ti][tj][r] = -1e30f;
                    }
        }

        // ---- online softmax (rows live in 16-lane halves) ----
#pragma unroll
        for (int ti = 0; ti < 2; ++ti)
#pragma unroll
            for (int r = 0; r < 8; ++r) {
                float mt = fmaxf(s[ti][0][r], s[ti][1][r]);
#pragma unroll
                for (int off = 1; off < 16; off <<= 1)
                    mt = fmaxf(mt, __shfl_xor(mt, off, 32));
                float mold = mrun[ti][r];
                float mnew = fmaxf(mold, mt);
                float rsc  = __expf(mold - mnew);
                srun[ti][r] *= rsc;
#pragma unroll
                for (int nj = 0; nj < 4; ++nj) yac[ti][nj][r] *= rsc;
                float ps = 0.f;
#pragma unroll
                for (int tj = 0; tj < 2; ++tj) {
                    float e = __expf(s[ti][tj][r] - mnew);
                    s[ti][tj][r] = e;
                    ps += e;
                }
#pragma unroll
                for (int off = 1; off < 16; off <<= 1)
                    ps += __shfl_xor(ps, off, 32);
                srun[ti][r] += ps;
                mrun[ti][r] = mnew;
            }

        // ---- P -> LDS bf16 [m][k], reload as A fragments ----
#pragma unroll
        for (int ti = 0; ti < 2; ++ti)
#pragma unroll
            for (int tj = 0; tj < 2; ++tj)
#pragma unroll
                for (int r = 0; r < 8; ++r) {
                    int mloc = ti * 16 + half * 8 + r;
                    int nloc = tj * 16 + m16;
                    lp[w][mloc * PP + nloc] = bf1(s[ti][tj][r]);
                }
        v16bf pf[2];
#pragma unroll
        for (int ti = 0; ti < 2; ++ti) {
            int base = (ti * 16 + m16) * PP + 8 * half;
            pf[ti] = ld_frag(&lp[w][base], &lp[w][base + 16]);
        }
        v16bf vf[4];
#pragma unroll
        for (int nj = 0; nj < 4; ++nj) {
            int base = (nj * 16 + m16) * VP + 8 * half;
            vf[nj] = ld_frag(&lv[w][base], &lv[w][base + 16]);
        }
#pragma unroll
        for (int ti = 0; ti < 2; ++ti)
#pragma unroll
            for (int nj = 0; nj < 4; ++nj)
                yac[ti][nj] = __builtin_amdgcn_wmma_f32_16x16x32_bf16(
                    false, pf[ti], false, vf[nj], (short)0, yac[ti][nj], false, false);
    }

    // ---- write per-wave partials ----
#pragma unroll
    for (int ti = 0; ti < 2; ++ti)
#pragma unroll
        for (int nj = 0; nj < 4; ++nj)
#pragma unroll
            for (int r = 0; r < 8; ++r)
                sY[w][ti * 16 + half * 8 + r][nj * 16 + m16] = yac[ti][nj][r];
    if (m16 == 0) {
#pragma unroll
        for (int ti = 0; ti < 2; ++ti)
#pragma unroll
            for (int r = 0; r < 8; ++r) {
                sM[w][ti * 16 + half * 8 + r] = mrun[ti][r];
                sS[w][ti * 16 + half * 8 + r] = srun[ti][r];
            }
    }
    __syncthreads();

    // ---- flash combine across 4 waves ----
    if (tid < QT) {
        float M = fmaxf(fmaxf(sM[0][tid], sM[1][tid]), fmaxf(sM[2][tid], sM[3][tid]));
        float den = 0.f;
#pragma unroll
        for (int ww = 0; ww < 4; ++ww) {
            float f = __expf(sM[ww][tid] - M);
            sFac[ww][tid] = f;
            den += sS[ww][tid] * f;
        }
        sDinv[tid] = 1.f / den;
    }
    __syncthreads();

    for (int c = tid; c < QT * DH; c += 128) {
        int m = c >> 6, d = c & 63;
        float v = sY[0][m][d] * sFac[0][m] + sY[1][m][d] * sFac[1][m]
                + sY[2][m][d] * sFac[2][m] + sY[3][m][d] * sFac[3][m];
        y[(size_t)(b * SEQ + qt * QT + m) * CH + hh * DH + d] = v * sDinv[m];
    }
}

// ---------------- host-side dispatch ----------------
static void run_gemm(const float* A, const float* Bm, const float* bias, const float* res,
                     float* out, int M, int N, int K, bool bt, bool gelu, hipStream_t s)
{
    dim3 grid((N + TN - 1) / TN, M / TM);
    if (bt)        gemm_wmma_bf16<true,  false><<<grid, 256, 0, s>>>(A, Bm, bias, res, out, M, N, K);
    else if (gelu) gemm_wmma_bf16<false, true ><<<grid, 256, 0, s>>>(A, Bm, bias, res, out, M, N, K);
    else           gemm_wmma_bf16<false, false><<<grid, 256, 0, s>>>(A, Bm, bias, res, out, M, N, K);
}

extern "C" void kernel_launch(void* const* d_in, const int* in_sizes, int n_in,
                              void* d_out, int out_size, void* d_ws, size_t ws_size,
                              hipStream_t stream)
{
    const int M = BAT * SEQ;  // 4096 rows

    const int*   idx    = (const int*)  d_in[0];
    const float* wte    = (const float*)d_in[1];
    const float* wpe    = (const float*)d_in[2];
    const float* ln1_w  = (const float*)d_in[3];
    const float* ln1_b  = (const float*)d_in[4];
    const float* ln2_w  = (const float*)d_in[5];
    const float* ln2_b  = (const float*)d_in[6];
    const float* attn_w = (const float*)d_in[7];
    const float* attn_b = (const float*)d_in[8];
    const float* proj_w = (const float*)d_in[9];
    const float* proj_b = (const float*)d_in[10];
    const float* fc_w   = (const float*)d_in[11];
    const float* fc_b   = (const float*)d_in[12];
    const float* fcp_w  = (const float*)d_in[13];
    const float* fcp_b  = (const float*)d_in[14];
    const float* lnf_w  = (const float*)d_in[15];
    const float* lnf_b  = (const float*)d_in[16];
    const float* lm     = (const float*)d_in[17];
    float* out = (float*)d_out;

    // workspace layout (fp32): x, h, qkv, y, act4  (~126 MB total)
    float* x    = (float*)d_ws;
    float* h    = x    + (size_t)M * CH;
    float* qkv  = h    + (size_t)M * CH;
    float* y    = qkv  + (size_t)M * 3 * CH;
    float* act4 = y    + (size_t)M * CH;

    embed_k<<<M, 256, 0, stream>>>(idx, wte, wpe, x);

    for (int l = 0; l < LYR; ++l) {
        const float* aw  = attn_w + (size_t)l * CH * 3 * CH;
        const float* ab  = attn_b + (size_t)l * 3 * CH;
        const float* pw  = proj_w + (size_t)l * CH * CH;
        const float* pb  = proj_b + (size_t)l * CH;
        const float* fw  = fc_w   + (size_t)l * CH * 4 * CH;
        const float* fb  = fc_b   + (size_t)l * 4 * CH;
        const float* fpw = fcp_w  + (size_t)l * 4 * CH * CH;
        const float* fpb = fcp_b  + (size_t)l * CH;

        layernorm_k<<<M, 256, 0, stream>>>(x, ln1_w + (size_t)l * CH, ln1_b + (size_t)l * CH, h);
        run_gemm(h, aw, ab, nullptr, qkv, M, 3 * CH, CH, false, false, stream);
        attn_wmma_k<<<dim3(SEQ / QT, BAT * HDS), 128, 0, stream>>>(qkv, y);
        run_gemm(y, pw, pb, x, x, M, CH, CH, false, false, stream);
        layernorm_k<<<M, 256, 0, stream>>>(x, ln2_w + (size_t)l * CH, ln2_b + (size_t)l * CH, h);
        run_gemm(h, fw, fb, nullptr, act4, M, 4 * CH, CH, false, true, stream);
        run_gemm(act4, fpw, fpb, x, x, M, CH, 4 * CH, false, false, stream);
    }

    layernorm_k<<<M, 256, 0, stream>>>(x, lnf_w, lnf_b, h);
    // logits = h @ lm_head^T  (lm_head is [V, C] row-major -> BT path), ragged N
    run_gemm(h, lm, nullptr, nullptr, out, M, VOC, CH, true, false, stream);
}
